// _RoIOffsetPooling_64622077935890
// MI455X (gfx1250) — matrix operand order repaired and checked by reading, not compiled, not run on hardware
//
#include <hip/hip_runtime.h>
#include <hip/hip_bf16.h>
#include <math.h>

// ---------------- problem constants ----------------
#define BB      2
#define CC      1024
#define HH      40
#define WW      40
#define SP      1600            // H*W
#define NN      3200            // B*H*W  (GEMM N)
#define KK      9216            // C*9    (GEMM K)
#define NROIS   64
#define SCALE   0.0625f

// ---------------- workspace layout (bytes, all 256-aligned) ----------------
static const size_t FEATT_OFF = 0;
static const size_t FEATT_SZ  = (size_t)BB * SP * CC * 4;        // 13,107,200
static const size_t WPT_OFF   = FEATT_OFF + FEATT_SZ;
static const size_t WPT_SZ    = (size_t)18 * 9 * CC * 4;         // 663,552
static const size_t AMAT_OFF  = WPT_OFF + WPT_SZ;
static const size_t AMAT_SZ   = (size_t)CC * KK * 2;             // 18,874,368
static const size_t OFFS_OFF  = AMAT_OFF + AMAT_SZ;
static const size_t OFFS_SZ   = (size_t)BB * 18 * SP * 4;        // 230,400
static const size_t BMAT_OFF  = OFFS_OFF + OFFS_SZ;
static const size_t BMAT_SZ   = (size_t)NN * KK * 2;             // 58,982,400
static const size_t DMAT_OFF  = BMAT_OFF + BMAT_SZ;
static const size_t DMAT_SZ   = (size_t)CC * NN * 4;             // 13,107,200

// ---------------- WMMA types ----------------
typedef __attribute__((ext_vector_type(16))) _Float16 v16h;
typedef __attribute__((ext_vector_type(8)))  float    v8f;
typedef __attribute__((ext_vector_type(4)))  _Float16 v4h;

union FragH { uint4 u[2]; v16h h; };

__device__ __forceinline__ v8f wmma_f16(const FragH& a, const FragH& b, v8f c) {
  return __builtin_amdgcn_wmma_f32_16x16x32_f16(false, a.h, false, b.h,
                                                (short)0, c, false, false);
}

// ============ K0: features [B,C,H,W] -> featT [B,H,W,C] ============
__global__ __launch_bounds__(256) void k_transpose_feat(const float* __restrict__ feat,
                                                        float* __restrict__ featT) {
  size_t tid = (size_t)blockIdx.x * 256 + threadIdx.x;       // B*SP*C total
  if (tid >= (size_t)BB * SP * CC) return;
  int c  = (int)(tid % CC);
  size_t r = tid / CC;                                       // b*SP + hw
  int hw = (int)(r % SP);
  int b  = (int)(r / SP);
  featT[tid] = feat[((size_t)(b * CC + c)) * SP + hw];
}

// ============ K1: w_pred [18,C,3,3] -> wpT [18,3,3,C] ============
__global__ __launch_bounds__(256) void k_transpose_wpred(const float* __restrict__ wp,
                                                         float* __restrict__ wpT) {
  int tid = blockIdx.x * 256 + threadIdx.x;                  // 18*9*C total
  if (tid >= 18 * 9 * CC) return;
  int c  = tid % CC;
  int t  = (tid / CC) % 9;
  int co = tid / (9 * CC);
  wpT[tid] = wp[((size_t)(co * CC + c)) * 9 + t];
}

// ============ K2: w_dconv [C,C,3,3] -> Amat f16 [Cout][t*1024+c] ============
__global__ __launch_bounds__(256) void k_make_A(const float* __restrict__ wd,
                                                _Float16* __restrict__ Am) {
  size_t tid = (size_t)blockIdx.x * 256 + threadIdx.x;       // C*KK total
  if (tid >= (size_t)CC * KK) return;
  int c  = (int)(tid % CC);
  int t  = (int)((tid / CC) % 9);
  int co = (int)(tid / KK);
  Am[tid] = (_Float16)wd[((size_t)(co * CC + c)) * 9 + t];
}

// ============ K3: offset-prediction 3x3 conv (SAME, zero pad) ============
__global__ __launch_bounds__(256) void k_offset_conv(const float* __restrict__ featT,
                                                     const float* __restrict__ wpT,
                                                     const float* __restrict__ bias,
                                                     float* __restrict__ offs) {
  int tid = blockIdx.x * 256 + threadIdx.x;                  // B*18*SP total
  if (tid >= BB * 18 * SP) return;
  int w  = tid % WW;
  int h  = (tid / WW) % HH;
  int co = (tid / SP) % 18;
  int b  = tid / (18 * SP);
  float acc = bias[co];
  for (int ky = 0; ky < 3; ++ky) {
    int y = h + ky - 1;
    if ((unsigned)y >= (unsigned)HH) continue;
    for (int kx = 0; kx < 3; ++kx) {
      int x = w + kx - 1;
      if ((unsigned)x >= (unsigned)WW) continue;
      const float* f  = featT + ((size_t)((b * HH + y) * WW + x)) * CC;
      const float* wv = wpT   + ((size_t)((co * 3 + ky) * 3 + kx)) * CC;
      float s = 0.f;
      #pragma unroll 4
      for (int c = 0; c < CC; c += 4) {
        float4 fv = *(const float4*)(f + c);
        float4 ww = *(const float4*)(wv + c);
        s = fmaf(fv.x, ww.x, s); s = fmaf(fv.y, ww.y, s);
        s = fmaf(fv.z, ww.z, s); s = fmaf(fv.w, ww.w, s);
      }
      acc += s;
    }
  }
  offs[tid] = acc;                                           // [B,18,H,W]
}

// ============ K4: bilinear sampling -> Bmat f16 [n][t*1024+c] ============
__global__ __launch_bounds__(256) void k_build_B(const float* __restrict__ featT,
                                                 const float* __restrict__ offs,
                                                 _Float16* __restrict__ Bm) {
  int blk = blockIdx.x;                                      // n*9 + t, 28800 blocks
  int t = blk % 9;
  int n = blk / 9;
  int b  = n / SP;
  int hw = n % SP;
  int h  = hw / WW;
  int w  = hw % WW;
  float dy = offs[((size_t)(b * 18 + 2 * t))     * SP + hw];
  float dx = offs[((size_t)(b * 18 + 2 * t + 1)) * SP + hw];
  float py = (float)h + (float)(t / 3 - 1) + dy;
  float px = (float)w + (float)(t % 3 - 1) + dx;
  float y0 = floorf(py), x0 = floorf(px);
  float wy1 = py - y0, wx1 = px - x0;
  float wy0 = 1.f - wy1, wx0 = 1.f - wx1;
  float ys[2] = { y0, y0 + 1.f }, xs[2] = { x0, x0 + 1.f };
  float wy[2] = { wy0, wy1 },     wx[2] = { wx0, wx1 };
  int   cidx[4];
  float cw[4];
  #pragma unroll
  for (int iy = 0; iy < 2; ++iy)
    #pragma unroll
    for (int ix = 0; ix < 2; ++ix) {
      float yf = ys[iy], xf = xs[ix];
      bool valid = (yf >= 0.f) & (yf <= (float)(HH - 1)) &
                   (xf >= 0.f) & (xf <= (float)(WW - 1));
      int yi = (int)fminf(fmaxf(yf, 0.f), (float)(HH - 1));
      int xi = (int)fminf(fmaxf(xf, 0.f), (float)(WW - 1));
      cidx[iy * 2 + ix] = ((b * HH + yi) * WW + xi) * CC;
      cw[iy * 2 + ix]   = valid ? wy[iy] * wx[ix] : 0.f;
    }
  int c0 = threadIdx.x * 4;                                  // 256 threads * 4 = 1024
  float4 v0 = *(const float4*)(featT + cidx[0] + c0);
  float4 v1 = *(const float4*)(featT + cidx[1] + c0);
  float4 v2 = *(const float4*)(featT + cidx[2] + c0);
  float4 v3 = *(const float4*)(featT + cidx[3] + c0);
  v4h o;
  o[0] = (_Float16)(v0.x * cw[0] + v1.x * cw[1] + v2.x * cw[2] + v3.x * cw[3]);
  o[1] = (_Float16)(v0.y * cw[0] + v1.y * cw[1] + v2.y * cw[2] + v3.y * cw[3]);
  o[2] = (_Float16)(v0.z * cw[0] + v1.z * cw[1] + v2.z * cw[2] + v3.z * cw[3]);
  o[3] = (_Float16)(v0.w * cw[0] + v1.w * cw[1] + v2.w * cw[2] + v3.w * cw[3]);
  *(v4h*)(Bm + (size_t)n * KK + t * CC + c0) = o;
}

// ============ K5: WMMA GEMM  D[1024][3200] = A[1024][9216] * B[9216][3200] ============
// Block tile 64M x 128N, 8 waves (2M x 4N), wave tile 32x32 -> 4 WMMAs per K-step.
// Double-buffered LDS + register-staged global loads: 1 barrier per K-step,
// global_load_b128s in flight across the WMMAs.
#define LDSTR 40   // padded f16 row stride (32 K + 8 pad) to spread LDS banks
__global__ __launch_bounds__(256) void k_gemm_wmma(const _Float16* __restrict__ A,
                                                   const _Float16* __restrict__ Bm,
                                                   float* __restrict__ D) {
  __shared__ _Float16 As[2][64 * LDSTR];
  __shared__ _Float16 Bs[2][128 * LDSTR];
  const int mTile = (blockIdx.x % (CC / 64)) * 64;           // 16 M tiles
  const int nTile = (blockIdx.x / (CC / 64)) * 128;          // 25 N tiles
  const int tid  = threadIdx.x;
  const int lane = tid & 31;
  const int wv   = tid >> 5;
  const int wm   = (wv & 1) * 32;                            // wave M offset in block
  const int wn   = (wv >> 1) * 32;                           // wave N offset in block
  const int lrow = tid >> 2;                                 // 0..63: LDS staging row
  const int lcol = (tid & 3) * 8;                            // f16 offset in 32-wide K row
  const int fm   = lane & 15;                                // fragment row/col id
  const int g    = lane >> 4;                                // lane half (ISA layouts)

  const _Float16* aG  = A  + (size_t)(mTile + lrow) * KK + lcol;
  const _Float16* bG0 = Bm + (size_t)(nTile + lrow) * KK + lcol;
  const _Float16* bG1 = Bm + (size_t)(nTile + 64 + lrow) * KK + lcol;

  // register-staged first tile
  uint4 ra  = *(const uint4*)(aG);
  uint4 rb0 = *(const uint4*)(bG0);
  uint4 rb1 = *(const uint4*)(bG1);

  v8f acc00 = {}, acc01 = {}, acc10 = {}, acc11 = {};
  int cur = 0;

  for (int k0 = 0; k0 < KK; k0 += 32) {
    *(uint4*)(&As[cur][lrow * LDSTR + lcol])        = ra;
    *(uint4*)(&Bs[cur][lrow * LDSTR + lcol])        = rb0;
    *(uint4*)(&Bs[cur][(64 + lrow) * LDSTR + lcol]) = rb1;
    __syncthreads();

    const int kn = k0 + 32;
    if (kn < KK) {                                           // loads fly over the WMMAs
      ra  = *(const uint4*)(aG  + kn);
      rb0 = *(const uint4*)(bG0 + kn);
      rb1 = *(const uint4*)(bG1 + kn);
      __builtin_prefetch(aG  + kn + 32, 0, 1);               // global_prefetch_b8
      __builtin_prefetch(bG0 + kn + 32, 0, 1);
      __builtin_prefetch(bG1 + kn + 32, 0, 1);
    }

    const _Float16* as = As[cur];
    const _Float16* bs = Bs[cur];
    // A fragments (ISA 16-bit A 16x32 layout): lane half g -> K {8g..+7, 16+8g..+7}
    FragH a0, a1, b0, b1;
    a0.u[0] = *(const uint4*)(as + (wm + fm) * LDSTR + 8 * g);
    a0.u[1] = *(const uint4*)(as + (wm + fm) * LDSTR + 16 + 8 * g);
    a1.u[0] = *(const uint4*)(as + (wm + 16 + fm) * LDSTR + 8 * g);
    a1.u[1] = *(const uint4*)(as + (wm + 16 + fm) * LDSTR + 16 + 8 * g);
    // B fragments (32x16): lane = N column, K {8g..+7, 16+8g..+7} (matches A)
    b0.u[0] = *(const uint4*)(bs + (wn + fm) * LDSTR + 8 * g);
    b0.u[1] = *(const uint4*)(bs + (wn + fm) * LDSTR + 16 + 8 * g);
    b1.u[0] = *(const uint4*)(bs + (wn + 16 + fm) * LDSTR + 8 * g);
    b1.u[1] = *(const uint4*)(bs + (wn + 16 + fm) * LDSTR + 16 + 8 * g);

    acc00 = wmma_f16(a0, b0, acc00);
    acc01 = wmma_f16(a0, b1, acc01);
    acc10 = wmma_f16(a1, b0, acc10);
    acc11 = wmma_f16(a1, b1, acc11);
    cur ^= 1;
  }

  // C/D layout: lane n = lane&15, VGPR v -> row v + 8*(lane>>4)
  const int col0 = nTile + wn + fm;
  #pragma unroll
  for (int v = 0; v < 8; ++v) {
    int row0 = mTile + wm + v + 8 * g;
    int row1 = row0 + 16;
    D[(size_t)row0 * NN + col0]      = acc00[v];
    D[(size_t)row0 * NN + col0 + 16] = acc01[v];
    D[(size_t)row1 * NN + col0]      = acc10[v];
    D[(size_t)row1 * NN + col0 + 16] = acc11[v];
  }
}

// ============ K6: Caffe RoIPool on Dmat [C][B*H*W] -> out [64,1024,7,7] ============
__global__ __launch_bounds__(256) void k_roi_pool(const float* __restrict__ D,
                                                  const float* __restrict__ rois,
                                                  float* __restrict__ out) {
  int tid = blockIdx.x * 256 + threadIdx.x;                  // NROIS*C*49 total
  if (tid >= NROIS * CC * 49) return;
  int j = tid % 7;
  int i = (tid / 7) % 7;
  int c = (tid / 49) % CC;
  int r = tid / (49 * CC);
  const float* rr = rois + r * 5;
  int   b    = (int)rr[0];
  float rs_w = roundf(rr[1] * SCALE);
  float rs_h = roundf(rr[2] * SCALE);
  float re_w = roundf(rr[3] * SCALE);
  float re_h = roundf(rr[4] * SCALE);
  float bin_h = fmaxf(re_h - rs_h + 1.f, 1.f) * (1.f / 7.f);
  float bin_w = fmaxf(re_w - rs_w + 1.f, 1.f) * (1.f / 7.f);
  int hs = (int)fminf(fmaxf(floorf((float)i * bin_h) + rs_h, 0.f), (float)HH);
  int he = (int)fminf(fmaxf(ceilf((float)(i + 1) * bin_h) + rs_h, 0.f), (float)HH);
  int ws = (int)fminf(fmaxf(floorf((float)j * bin_w) + rs_w, 0.f), (float)WW);
  int we = (int)fminf(fmaxf(ceilf((float)(j + 1) * bin_w) + rs_w, 0.f), (float)WW);
  float m = -INFINITY;
  const float* base = D + (size_t)c * NN + b * SP;
  for (int y = hs; y < he; ++y)
    for (int x = ws; x < we; ++x)
      m = fmaxf(m, base[y * WW + x]);
  out[tid] = (m == -INFINITY) ? 0.f : m;                     // empty bin -> 0
}

// ---------------------------------------------------------------------------
extern "C" void kernel_launch(void* const* d_in, const int* in_sizes, int n_in,
                              void* d_out, int out_size, void* d_ws, size_t ws_size,
                              hipStream_t stream) {
  const float* features = (const float*)d_in[0];   // [2,1024,40,40]
  const float* rois     = (const float*)d_in[1];   // [64,5]
  const float* w_pred   = (const float*)d_in[2];   // [18,1024,3,3]
  const float* b_pred   = (const float*)d_in[3];   // [18]
  const float* w_dconv  = (const float*)d_in[4];   // [1024,1024,3,3]
  (void)in_sizes; (void)n_in; (void)out_size; (void)ws_size;

  char* ws = (char*)d_ws;
  float*    featT = (float*)   (ws + FEATT_OFF);
  float*    wpT   = (float*)   (ws + WPT_OFF);
  _Float16* Amat  = (_Float16*)(ws + AMAT_OFF);
  float*    offs  = (float*)   (ws + OFFS_OFF);
  _Float16* Bmat  = (_Float16*)(ws + BMAT_OFF);
  float*    Dmat  = (float*)   (ws + DMAT_OFF);
  float*    outp  = (float*)d_out;

  // K0: channel-last features
  k_transpose_feat<<<(BB * SP * CC + 255) / 256, 256, 0, stream>>>(features, featT);
  // K1: transposed offset-conv weights
  k_transpose_wpred<<<(18 * 9 * CC + 255) / 256, 256, 0, stream>>>(w_pred, wpT);
  // K2: f16 GEMM A matrix from w_dconv
  k_make_A<<<(int)(((size_t)CC * KK + 255) / 256), 256, 0, stream>>>(w_dconv, Amat);
  // K3: offsets
  k_offset_conv<<<(BB * 18 * SP + 255) / 256, 256, 0, stream>>>(featT, wpT, b_pred, offs);
  // K4: bilinear-sampled f16 GEMM B matrix
  k_build_B<<<NN * 9, 256, 0, stream>>>(featT, offs, Bmat);
  // K5: WMMA GEMM (M=1024, N=3200, K=9216), block tile 64x128
  k_gemm_wmma<<<(CC / 64) * (NN / 128), 256, 0, stream>>>(Amat, Bmat, Dmat);
  // K6: RoI max pool
  k_roi_pool<<<(NROIS * CC * 49 + 255) / 256, 256, 0, stream>>>(Dmat, rois, outp);
}
